// OneBitLinear_88270167867535
// MI455X (gfx1250) — compile-verified, matrix operand order checked by reading, not run on hardware
//
#include <hip/hip_runtime.h>

typedef __attribute__((ext_vector_type(16))) __bf16 bf16x16;
typedef __attribute__((ext_vector_type(8)))  __bf16 bf16x8;
typedef __attribute__((ext_vector_type(4)))  __bf16 bf16x4;
typedef __attribute__((ext_vector_type(8)))  float  f32x8;

#define DIN   4096
#define DOUT  4096
#define MROWS 8192          // B*S = 4*2048
#define TILE_K 64
#define LDS_STRIDE 72       // 64 + 8 pad bf16 elems per row (144B, 16B aligned)
#define LN_EPS 1e-5f

// float -> bf16, round to nearest even
__device__ inline __bf16 f2bf(float f) {
    union { float f; unsigned u; } a; a.f = f;
    unsigned r = a.u + 0x7FFFu + ((a.u >> 16) & 1u);
    unsigned short h = (unsigned short)(r >> 16);
    union { unsigned short s; __bf16 b; } o; o.s = h;
    return o.b;
}

__device__ inline void wait_async0() {
#if __has_builtin(__builtin_amdgcn_s_wait_asynccnt)
    __builtin_amdgcn_s_wait_asynccnt(0);
#else
    asm volatile("s_wait_asynccnt 0x0" ::: "memory");
#endif
}

// async global -> LDS, 16B per lane, ASYNCcnt-tracked (GV addressing mode)
__device__ inline void async_copy_b128(unsigned lds_addr, const void* gaddr) {
    asm volatile("global_load_async_to_lds_b128 %0, %1, off"
                 :: "v"(lds_addr), "v"(gaddr) : "memory");
}

__device__ inline unsigned lds_offset_of(const void* shared_ptr) {
    // LDS aperture keeps the wave-relative LDS byte offset in addr[31:0]
    return (unsigned)(size_t)shared_ptr;
}

// ---------------- kernel 1: xs[m][k] = bf16(x[m][k] * in_scale[k]) ----------
__global__ __launch_bounds__(256)
void quant_x_kernel(const float* __restrict__ x,
                    const float* __restrict__ in_scale,
                    __bf16* __restrict__ xs) {
    long idx  = (long)blockIdx.x * blockDim.x + threadIdx.x;  // 4 elems/thread
    long base = idx * 4;
    int  k    = (int)(base & (long)(DIN - 1));
    float4 v = *(const float4*)(x + base);
    float4 s = *(const float4*)(in_scale + k);
    bf16x4 r;
    r[0] = f2bf(v.x * s.x);
    r[1] = f2bf(v.y * s.y);
    r[2] = f2bf(v.z * s.z);
    r[3] = f2bf(v.w * s.w);
    *(bf16x4*)(xs + base) = r;
}

// ---------------- kernel 2: wb[o][k] = bf16(sign(w[o][k])) ------------------
__global__ __launch_bounds__(256)
void quant_w_kernel(const float* __restrict__ w,
                    __bf16* __restrict__ wb) {
    long idx  = (long)blockIdx.x * blockDim.x + threadIdx.x;
    long base = idx * 4;
    float4 v = *(const float4*)(w + base);
    bf16x4 r;
    r[0] = f2bf((v.x > 0.f) ? 1.f : ((v.x < 0.f) ? -1.f : 0.f));
    r[1] = f2bf((v.y > 0.f) ? 1.f : ((v.y < 0.f) ? -1.f : 0.f));
    r[2] = f2bf((v.z > 0.f) ? 1.f : ((v.z < 0.f) ? -1.f : 0.f));
    r[3] = f2bf((v.w > 0.f) ? 1.f : ((v.w < 0.f) ? -1.f : 0.f));
    *(bf16x4*)(wb + base) = r;
}

// ---------------- kernel 3: WMMA GEMM, out = xs @ wb^T * out_scale + bias ---
// Block: 256 threads = 8 waves in 4(M) x 2(N) grid. Block tile: 256(M) x 128(N).
// Wave tile: 64x64 = 4x4 fragments of 16x16 -> 16 WMMA per 8 fragment loads.
// K staged 64 at a time, double-buffered LDS filled by async-to-LDS copies.
__global__ __launch_bounds__(256)
void gemm_bf16_kernel(const __bf16* __restrict__ xs,    // [MROWS][DIN]
                      const __bf16* __restrict__ wb,    // [DOUT][DIN]
                      const float*  __restrict__ out_scale,
                      const float*  __restrict__ bias,
                      float* __restrict__ out) {        // [MROWS][DOUT]
    __shared__ __align__(16) __bf16 ldsA[2][256 * LDS_STRIDE];  // 2 x 36 KB
    __shared__ __align__(16) __bf16 ldsB[2][128 * LDS_STRIDE];  // 2 x 18 KB

    const int t     = threadIdx.x;
    const int lane  = t & 31;
    const int wave  = t >> 5;        // 0..7
    const int waveM = wave >> 1;     // 0..3
    const int waveN = wave & 1;      // 0..1
    const int m0 = blockIdx.y * 256;
    const int n0 = blockIdx.x * 128;
    const int half = lane >> 4;      // 0/1 (K-half select per WMMA layout)
    const int l15  = lane & 15;

    const f32x8 vzero = {0.f, 0.f, 0.f, 0.f, 0.f, 0.f, 0.f, 0.f};
    f32x8 acc[4][4];
#pragma unroll
    for (int i = 0; i < 4; ++i)
#pragma unroll
        for (int j = 0; j < 4; ++j) acc[i][j] = vzero;

    // async fill of (buf, k0): A = 256x64 bf16 (32KB), B = 128x64 bf16 (16KB)
    auto issue_tile = [&](int buf, int k0) {
#pragma unroll
        for (int i = 0; i < 8; ++i) {          // A tile: 2048 b128 transfers
            int idx = t + i * 256;             // 0..2047
            int row = idx >> 3;                // 0..255
            int kc  = (idx & 7) * 8;           // 0..56
            async_copy_b128(lds_offset_of(&ldsA[buf][row * LDS_STRIDE + kc]),
                            xs + (long)(m0 + row) * DIN + k0 + kc);
        }
#pragma unroll
        for (int i = 0; i < 4; ++i) {          // B tile: 1024 b128 transfers
            int idx = t + i * 256;             // 0..1023
            int row = idx >> 3;                // 0..127
            int kc  = (idx & 7) * 8;           // 0..56
            async_copy_b128(lds_offset_of(&ldsB[buf][row * LDS_STRIDE + kc]),
                            wb + (long)(n0 + row) * DIN + k0 + kc);
        }
    };

    issue_tile(0, 0);
    wait_async0();
    __syncthreads();

    int buf = 0;
    for (int k0 = 0; k0 < DIN; k0 += TILE_K) {
        // overlap: start filling the other buffer while computing this one
        if (k0 + TILE_K < DIN) issue_tile(buf ^ 1, k0 + TILE_K);

        const __bf16* bA = &ldsA[buf][0];
        const __bf16* bB = &ldsB[buf][0];

#pragma unroll
        for (int kk = 0; kk < TILE_K; kk += 32) {
            // load all 8 fragments for this K-step, then run 16 WMMAs
            bf16x16 afrag[4];
            bf16x16 bfrag[4];
#pragma unroll
            for (int fn = 0; fn < 4; ++fn) {
                int row = waveN * 64 + fn * 16 + l15;
                bf16x8 lo = *(const bf16x8*)&bB[row * LDS_STRIDE + kk + half * 8];
                bf16x8 hi = *(const bf16x8*)&bB[row * LDS_STRIDE + kk + half * 8 + 16];
#pragma unroll
                for (int e = 0; e < 8; ++e) { bfrag[fn][e] = lo[e]; bfrag[fn][8 + e] = hi[e]; }
            }
#pragma unroll
            for (int fm = 0; fm < 4; ++fm) {
                int row = waveM * 64 + fm * 16 + l15;
                bf16x8 lo = *(const bf16x8*)&bA[row * LDS_STRIDE + kk + half * 8];
                bf16x8 hi = *(const bf16x8*)&bA[row * LDS_STRIDE + kk + half * 8 + 16];
#pragma unroll
                for (int e = 0; e < 8; ++e) { afrag[fm][e] = lo[e]; afrag[fm][8 + e] = hi[e]; }
            }
#pragma unroll
            for (int fm = 0; fm < 4; ++fm)
#pragma unroll
                for (int fn = 0; fn < 4; ++fn) {
                    acc[fm][fn] = __builtin_amdgcn_wmma_f32_16x16x32_bf16(
                        false, afrag[fm], false, bfrag[fn],
                        (short)0, acc[fm][fn], false, false);
                }
        }

        // own async fills done, then workgroup-wide visibility for next iter
        wait_async0();
        __syncthreads();
        buf ^= 1;
    }

    // ---- epilogue: out_scale * acc + bias, fp32 store ----
    // C layout: VGPR v -> M = v + 8*(lane>=16), N = lane&15
#pragma unroll
    for (int fn = 0; fn < 4; ++fn) {
        int n = n0 + waveN * 64 + fn * 16 + l15;
        float sc = out_scale[n];
        float bs = bias[n];
#pragma unroll
        for (int fm = 0; fm < 4; ++fm) {
            int mb = m0 + waveM * 64 + fm * 16 + half * 8;
#pragma unroll
            for (int v = 0; v < 8; ++v) {
                out[(long)(mb + v) * DOUT + n] = acc[fm][fn][v] * sc + bs;
            }
        }
    }
}

// ---------------- kernel 4: row-wise LayerNorm (in place on out) ------------
__global__ __launch_bounds__(256)
void ln_kernel(float* __restrict__ h,
               const float* __restrict__ gamma,
               const float* __restrict__ beta) {
    __shared__ float red[256];
    const int t = threadIdx.x;
    float* p = h + (long)blockIdx.x * DOUT;

    float vals[DOUT / 256];
    float sum = 0.f;
#pragma unroll
    for (int i = 0; i < DOUT / 256; ++i) {
        vals[i] = p[t + i * 256];
        sum += vals[i];
    }
    red[t] = sum;
    __syncthreads();
    for (int s = 128; s >= 1; s >>= 1) {
        if (t < s) red[t] += red[t + s];
        __syncthreads();
    }
    float mu = red[0] * (1.0f / DOUT);
    __syncthreads();

    float sq = 0.f;
#pragma unroll
    for (int i = 0; i < DOUT / 256; ++i) {
        float d = vals[i] - mu;
        sq += d * d;
    }
    red[t] = sq;
    __syncthreads();
    for (int s = 128; s >= 1; s >>= 1) {
        if (t < s) red[t] += red[t + s];
        __syncthreads();
    }
    float rstd = rsqrtf(red[0] * (1.0f / DOUT) + LN_EPS);

#pragma unroll
    for (int i = 0; i < DOUT / 256; ++i) {
        int c = t + i * 256;
        p[c] = (vals[i] - mu) * rstd * gamma[c] + beta[c];
    }
}

// ---------------------------------------------------------------------------
extern "C" void kernel_launch(void* const* d_in, const int* in_sizes, int n_in,
                              void* d_out, int out_size, void* d_ws, size_t ws_size,
                              hipStream_t stream) {
    (void)in_sizes; (void)n_in; (void)out_size; (void)ws_size;
    const float* x         = (const float*)d_in[0];
    const float* w         = (const float*)d_in[1];
    const float* in_scale  = (const float*)d_in[2];
    const float* out_scale = (const float*)d_in[3];
    const float* bias      = (const float*)d_in[4];
    const float* gamma     = (const float*)d_in[5];
    const float* beta      = (const float*)d_in[6];
    float* out = (float*)d_out;

    // workspace layout: xs bf16 [MROWS*DIN] (67MB), wb bf16 [DOUT*DIN] (33.5MB)
    __bf16* xs = (__bf16*)d_ws;
    __bf16* wb = xs + (size_t)MROWS * DIN;

    {   // xs = bf16(x * in_scale): 33.55M elems, 4 per thread
        long nthreads = (long)MROWS * DIN / 4;
        quant_x_kernel<<<(int)(nthreads / 256), 256, 0, stream>>>(x, in_scale, xs);
    }
    {   // wb = bf16(sign(w)): 16.78M elems, 4 per thread
        long nthreads = (long)DOUT * DIN / 4;
        quant_w_kernel<<<(int)(nthreads / 256), 256, 0, stream>>>(w, wb);
    }
    {
        dim3 grid(DOUT / 128, MROWS / 256);  // 32 x 32 blocks
        gemm_bf16_kernel<<<grid, 256, 0, stream>>>(xs, wb, out_scale, bias, out);
    }
    ln_kernel<<<MROWS, 256, 0, stream>>>(out, gamma, beta);
}
